// WienerHammersteinModel_82377472737777
// MI455X (gfx1250) — matrix operand order repaired and checked by reading, not compiled, not run on hardware
//
#include <hip/hip_runtime.h>
#include <math.h>

typedef __attribute__((ext_vector_type(16))) __bf16 v16bf;
typedef __attribute__((ext_vector_type(8)))  __bf16 v8bf;
typedef __attribute__((ext_vector_type(8)))  float  v8f;

// ---------------- problem constants ----------------
#define KBATCH   65536
#define KSD      256     // SD1 == SD2 == NL_IN == NL_OUT
#define KIN      128     // IN_D == OUT_D
#define KNH      1024    // MLP hidden
#define KROWS    128     // batch rows per workgroup (8 strips of 16)
#define KTHREADS 256     // 8 wave32: 4 strip-pairs x 2 column halves

// dynamic LDS: sX = KROWS*KSD bf16 (64KB), sScr = KROWS*KIN bf16 (32KB)
#define SMEM_ELEMS (KROWS * KSD + KROWS * KIN)
#define SMEM_BYTES (SMEM_ELEMS * 2)

// ---------------- bf16 weight arena offsets in d_ws (elements) ----------------
#define OFF_A1  0
#define OFF_B1  (OFF_A1 + KSD*KSD)   // 65536
#define OFF_C1  (OFF_B1 + KSD*KIN)   // 98304
#define OFF_D1  (OFF_C1 + KSD*KSD)   // 163840
#define OFF_A2  (OFF_D1 + KSD*KIN)   // 196608
#define OFF_B2  (OFF_A2 + KSD*KSD)   // 262144
#define OFF_C2  (OFF_B2 + KSD*KSD)   // 327680
#define OFF_D2  (OFF_C2 + KIN*KSD)   // 360448
#define OFF_W1  (OFF_D2 + KIN*KSD)   // 393216
#define OFF_W2  (OFF_W1 + KNH*KSD)   // 655360
#define W_TOTAL (OFF_W2 + KSD*KNH)   // 917504 bf16 (~1.75 MB of d_ws)

// ---------------- helpers ----------------
__device__ __forceinline__ v8f zero8() {
    v8f z;
#pragma unroll
    for (int i = 0; i < 8; ++i) z[i] = 0.0f;
    return z;
}

union BF16x16 { v16bf v; v8bf h[2]; };

// D = A(16x32 bf16) x B(32x16 bf16) + C(16x16 f32)
__device__ __forceinline__ v8f wmma_bf16(v16bf a, v16bf b, v8f c) {
    return __builtin_amdgcn_wmma_f32_16x16x32_bf16(
        /*neg_a=*/false, a, /*neg_b=*/false, b,
        /*c_mod=*/(short)0, c, /*reuse_a=*/false, /*reuse_b=*/false);
}

// A-fragment from row-major bf16 in LDS. ISA 7.12.2, 16-bit A 16x32:
// lane L: m = L&15, s = L>>4; halves 0..7 -> K = k0+s*8+h ; 8..15 -> K = k0+16+s*8+h
__device__ __forceinline__ v16bf load_a_lds(const __bf16* base, int ld, int m0, int k0, int lane) {
    const int m = m0 + (lane & 15);
    const int s = lane >> 4;
    BF16x16 u;
    u.h[0] = *(const v8bf*)(base + m * ld + k0 + s * 8);
    u.h[1] = *(const v8bf*)(base + m * ld + k0 + 16 + s * 8);
    return u.v;
}

// A-fragment from row-major f32 in global memory (convert on the fly).
__device__ __forceinline__ v16bf load_a_gf32(const float* base, int ld, int m0, int k0, int lane) {
    const int m = m0 + (lane & 15);
    const int s = lane >> 4;
    const float* p = base + (size_t)m * ld + k0 + s * 8;
    v16bf r;
#pragma unroll
    for (int i = 0; i < 8; ++i) r[i] = (__bf16)p[i];
#pragma unroll
    for (int i = 0; i < 8; ++i) r[i + 8] = (__bf16)p[16 + i];
    return r;
}

// B-fragment: Y = X @ W^T, W row-major [N,K] bf16 => B col n == W row n.
// lane L: n = n0+(L&15), s = L>>4; halves 0..7 -> K = k0+s*8+h ; 8..15 -> K = k0+16+s*8+h
__device__ __forceinline__ v16bf load_b_w(const __bf16* W, int ldw, int n0, int k0, int lane) {
    const int n = n0 + (lane & 15);
    const int s = lane >> 4;
    const __bf16* p = W + (size_t)n * ldw + k0 + s * 8;
    BF16x16 u;
    u.h[0] = *(const v8bf*)(p);
    u.h[1] = *(const v8bf*)(p + 16);
    return u.v;
}

// ---------------- weight convert kernel (f32 -> bf16 arena) ----------------
__global__ void wh_cvt_f32_bf16(const float* __restrict__ src, __bf16* __restrict__ dst, int n) {
    int i = blockIdx.x * blockDim.x + threadIdx.x;
    if (i < n) dst[i] = (__bf16)src[i];
}

// ---------------- fused Wiener-Hammerstein kernel ----------------
__global__ __launch_bounds__(KTHREADS)
void wh_fused(const float* __restrict__ s1g, const float* __restrict__ s2g,
              const float* __restrict__ ug,
              const float* __restrict__ b1, const float* __restrict__ b2,
              const __bf16* __restrict__ wbf,
              float* __restrict__ ons1, float* __restrict__ ons2, float* __restrict__ oy) {
    extern __shared__ __align__(16) __bf16 smem[];
    __bf16* sX   = smem;                 // [KROWS x 256]: state1 -> v -> w (64KB)
    __bf16* sScr = smem + KROWS * KSD;   // [KROWS x 128]: u ; later tanh tile [KROWS x 64]

    const int tid  = threadIdx.x;
    const int wave = tid >> 5;
    const int lane = tid & 31;
    const int row0 = blockIdx.x * KROWS;

    const __bf16* wA1 = wbf + OFF_A1;
    const __bf16* wB1 = wbf + OFF_B1;
    const __bf16* wC1 = wbf + OFF_C1;
    const __bf16* wD1 = wbf + OFF_D1;
    const __bf16* wA2 = wbf + OFF_A2;
    const __bf16* wB2 = wbf + OFF_B2;
    const __bf16* wC2 = wbf + OFF_C2;
    const __bf16* wD2 = wbf + OFF_D2;
    const __bf16* wW1 = wbf + OFF_W1;
    const __bf16* wW2 = wbf + OFF_W2;

    // ---- Phase A: stage state1 and u into LDS as bf16 ----
    for (int i = tid * 4; i < KROWS * KSD; i += KTHREADS * 4) {
        int r = i >> 8, c = i & (KSD - 1);
        float4 f = *(const float4*)(s1g + (size_t)(row0 + r) * KSD + c);
        sX[i + 0] = (__bf16)f.x; sX[i + 1] = (__bf16)f.y;
        sX[i + 2] = (__bf16)f.z; sX[i + 3] = (__bf16)f.w;
    }
    for (int i = tid * 4; i < KROWS * KIN; i += KTHREADS * 4) {
        int r = i >> 7, c = i & (KIN - 1);
        float4 f = *(const float4*)(ug + (size_t)(row0 + r) * KIN + c);
        sScr[i + 0] = (__bf16)f.x; sScr[i + 1] = (__bf16)f.y;
        sScr[i + 2] = (__bf16)f.z; sScr[i + 3] = (__bf16)f.w;
    }
    __syncthreads();

    // Wave tiling: 8 row strips as 4 strip-pairs; 2 waves per strip-pair (col halves).
    // Each B-fragment load feeds 2 WMMAs (both strips) -> halves L2 weight traffic.
    const int m0      = (wave >> 1) * 32;  // first strip of this wave's pair
    const int colhalf = wave & 1;          // low/high 128 output columns
    const int srow    = lane >> 4;         // D-layout row selector
    const int scol    = lane & 15;         // D-layout column

    // ---- Phase B: ns1 = s1@A1' + u@B1' -> global ----
    {
        v8f acc[2][8];
#pragma unroll
        for (int j = 0; j < 8; ++j) { acc[0][j] = zero8(); acc[1][j] = zero8(); }
        for (int k0 = 0; k0 < KSD; k0 += 32) {
            v16bf a0 = load_a_lds(sX, KSD, m0,      k0, lane);
            v16bf a1 = load_a_lds(sX, KSD, m0 + 16, k0, lane);
#pragma unroll
            for (int j = 0; j < 8; ++j) {
                int n0 = (colhalf * 8 + j) * 16;
                v16bf b = load_b_w(wA1, KSD, n0, k0, lane);
                acc[0][j] = wmma_bf16(a0, b, acc[0][j]);
                acc[1][j] = wmma_bf16(a1, b, acc[1][j]);
            }
        }
        for (int k0 = 0; k0 < KIN; k0 += 32) {
            v16bf a0 = load_a_lds(sScr, KIN, m0,      k0, lane);
            v16bf a1 = load_a_lds(sScr, KIN, m0 + 16, k0, lane);
#pragma unroll
            for (int j = 0; j < 8; ++j) {
                int n0 = (colhalf * 8 + j) * 16;
                v16bf b = load_b_w(wB1, KIN, n0, k0, lane);
                acc[0][j] = wmma_bf16(a0, b, acc[0][j]);
                acc[1][j] = wmma_bf16(a1, b, acc[1][j]);
            }
        }
#pragma unroll
        for (int st = 0; st < 2; ++st) {
#pragma unroll
            for (int j = 0; j < 8; ++j) {
                int n0 = (colhalf * 8 + j) * 16;
#pragma unroll
                for (int r = 0; r < 8; ++r) {
                    int m = m0 + st * 16 + r + 8 * srow;
                    ons1[(size_t)(row0 + m) * KSD + n0 + scol] = acc[st][j][r];
                }
            }
        }
    }

    // ---- Phase C: v = s1@C1' + u@D1' -> LDS (overwrites state1 tile) ----
    {
        v8f acc[2][8];
#pragma unroll
        for (int j = 0; j < 8; ++j) { acc[0][j] = zero8(); acc[1][j] = zero8(); }
        for (int k0 = 0; k0 < KSD; k0 += 32) {
            v16bf a0 = load_a_lds(sX, KSD, m0,      k0, lane);
            v16bf a1 = load_a_lds(sX, KSD, m0 + 16, k0, lane);
#pragma unroll
            for (int j = 0; j < 8; ++j) {
                int n0 = (colhalf * 8 + j) * 16;
                v16bf b = load_b_w(wC1, KSD, n0, k0, lane);
                acc[0][j] = wmma_bf16(a0, b, acc[0][j]);
                acc[1][j] = wmma_bf16(a1, b, acc[1][j]);
            }
        }
        for (int k0 = 0; k0 < KIN; k0 += 32) {
            v16bf a0 = load_a_lds(sScr, KIN, m0,      k0, lane);
            v16bf a1 = load_a_lds(sScr, KIN, m0 + 16, k0, lane);
#pragma unroll
            for (int j = 0; j < 8; ++j) {
                int n0 = (colhalf * 8 + j) * 16;
                v16bf b = load_b_w(wD1, KIN, n0, k0, lane);
                acc[0][j] = wmma_bf16(a0, b, acc[0][j]);
                acc[1][j] = wmma_bf16(a1, b, acc[1][j]);
            }
        }
        __syncthreads();  // all reads of sX/sScr complete before v overwrites sX
#pragma unroll
        for (int st = 0; st < 2; ++st) {
#pragma unroll
            for (int j = 0; j < 8; ++j) {
                int n0 = (colhalf * 8 + j) * 16;
#pragma unroll
                for (int r = 0; r < 8; ++r) {
                    int m = m0 + st * 16 + r + 8 * srow;
                    sX[m * KSD + n0 + scol] = (__bf16)acc[st][j][r];
                }
            }
        }
        __syncthreads();
    }

    // ---- Phase D: w = tanh(v@W1' + b1)@W2'; hidden activations never leave the WGP ----
    __bf16* sH = sScr;  // [KROWS x 64] tanh tile (16KB), aliases dead u buffer
    v8f wacc[2][8];
#pragma unroll
    for (int j = 0; j < 8; ++j) { wacc[0][j] = zero8(); wacc[1][j] = zero8(); }

    const int j2base = (wave & 1) * 2;  // 2 of the 4 hidden col-tiles per wave

    for (int hb = 0; hb < KNH; hb += 64) {
        // stage 1: h tile [KROWS x 64] = tanh(v @ W1[hb:hb+64]^T + b1)
        v8f hacc[2][2];
        hacc[0][0] = zero8(); hacc[0][1] = zero8();
        hacc[1][0] = zero8(); hacc[1][1] = zero8();
        for (int k0 = 0; k0 < KSD; k0 += 32) {
            v16bf a0 = load_a_lds(sX, KSD, m0,      k0, lane);
            v16bf a1 = load_a_lds(sX, KSD, m0 + 16, k0, lane);
#pragma unroll
            for (int j = 0; j < 2; ++j) {
                int n0 = (j2base + j) * 16;
                v16bf b = load_b_w(wW1, KSD, hb + n0, k0, lane);
                hacc[0][j] = wmma_bf16(a0, b, hacc[0][j]);
                hacc[1][j] = wmma_bf16(a1, b, hacc[1][j]);
            }
        }
#pragma unroll
        for (int st = 0; st < 2; ++st) {
#pragma unroll
            for (int j = 0; j < 2; ++j) {
                int n0 = (j2base + j) * 16;
                float bv = b1[hb + n0 + scol];
#pragma unroll
                for (int r = 0; r < 8; ++r) {
                    int m = m0 + st * 16 + r + 8 * srow;
                    sH[m * 64 + n0 + scol] = (__bf16)tanhf(hacc[st][j][r] + bv);
                }
            }
        }
        __syncthreads();
        // stage 2: wacc += h_tile @ W2[:, hb:hb+64]^T
        for (int k0 = 0; k0 < 64; k0 += 32) {
            v16bf a0 = load_a_lds(sH, 64, m0,      k0, lane);
            v16bf a1 = load_a_lds(sH, 64, m0 + 16, k0, lane);
#pragma unroll
            for (int j = 0; j < 8; ++j) {
                int n0 = (colhalf * 8 + j) * 16;
                v16bf b = load_b_w(wW2, KNH, n0, hb + k0, lane);
                wacc[0][j] = wmma_bf16(a0, b, wacc[0][j]);
                wacc[1][j] = wmma_bf16(a1, b, wacc[1][j]);
            }
        }
        __syncthreads();
    }

    // ---- Phase E: w = wacc + b2 -> LDS bf16 (overwrites v) ----
#pragma unroll
    for (int st = 0; st < 2; ++st) {
#pragma unroll
        for (int j = 0; j < 8; ++j) {
            int n0 = (colhalf * 8 + j) * 16;
            float bv = b2[n0 + scol];
#pragma unroll
            for (int r = 0; r < 8; ++r) {
                int m = m0 + st * 16 + r + 8 * srow;
                sX[m * KSD + n0 + scol] = (__bf16)(wacc[st][j][r] + bv);
            }
        }
    }
    __syncthreads();

    // ---- Phase G: ns2 = s2@A2' + w@B2' -> global ----
    const float* s2base = s2g + (size_t)row0 * KSD;
    {
        v8f acc[2][8];
#pragma unroll
        for (int j = 0; j < 8; ++j) { acc[0][j] = zero8(); acc[1][j] = zero8(); }
        for (int k0 = 0; k0 < KSD; k0 += 32) {
            v16bf a0 = load_a_gf32(s2base, KSD, m0,      k0, lane);
            v16bf a1 = load_a_gf32(s2base, KSD, m0 + 16, k0, lane);
#pragma unroll
            for (int j = 0; j < 8; ++j) {
                int n0 = (colhalf * 8 + j) * 16;
                v16bf b = load_b_w(wA2, KSD, n0, k0, lane);
                acc[0][j] = wmma_bf16(a0, b, acc[0][j]);
                acc[1][j] = wmma_bf16(a1, b, acc[1][j]);
            }
        }
        for (int k0 = 0; k0 < KSD; k0 += 32) {
            v16bf a0 = load_a_lds(sX, KSD, m0,      k0, lane);
            v16bf a1 = load_a_lds(sX, KSD, m0 + 16, k0, lane);
#pragma unroll
            for (int j = 0; j < 8; ++j) {
                int n0 = (colhalf * 8 + j) * 16;
                v16bf b = load_b_w(wB2, KSD, n0, k0, lane);
                acc[0][j] = wmma_bf16(a0, b, acc[0][j]);
                acc[1][j] = wmma_bf16(a1, b, acc[1][j]);
            }
        }
#pragma unroll
        for (int st = 0; st < 2; ++st) {
#pragma unroll
            for (int j = 0; j < 8; ++j) {
                int n0 = (colhalf * 8 + j) * 16;
#pragma unroll
                for (int r = 0; r < 8; ++r) {
                    int m = m0 + st * 16 + r + 8 * srow;
                    ons2[(size_t)(row0 + m) * KSD + n0 + scol] = acc[st][j][r];
                }
            }
        }
    }

    // ---- Phase H: y = s2@C2' + w@D2' -> global ([KROWS x 128]) ----
    {
        v8f acc[2][4];
#pragma unroll
        for (int j = 0; j < 4; ++j) { acc[0][j] = zero8(); acc[1][j] = zero8(); }
        for (int k0 = 0; k0 < KSD; k0 += 32) {
            v16bf a0 = load_a_gf32(s2base, KSD, m0,      k0, lane);
            v16bf a1 = load_a_gf32(s2base, KSD, m0 + 16, k0, lane);
#pragma unroll
            for (int j = 0; j < 4; ++j) {
                int n0 = (colhalf * 4 + j) * 16;
                v16bf b = load_b_w(wC2, KSD, n0, k0, lane);
                acc[0][j] = wmma_bf16(a0, b, acc[0][j]);
                acc[1][j] = wmma_bf16(a1, b, acc[1][j]);
            }
        }
        for (int k0 = 0; k0 < KSD; k0 += 32) {
            v16bf a0 = load_a_lds(sX, KSD, m0,      k0, lane);
            v16bf a1 = load_a_lds(sX, KSD, m0 + 16, k0, lane);
#pragma unroll
            for (int j = 0; j < 4; ++j) {
                int n0 = (colhalf * 4 + j) * 16;
                v16bf b = load_b_w(wD2, KSD, n0, k0, lane);
                acc[0][j] = wmma_bf16(a0, b, acc[0][j]);
                acc[1][j] = wmma_bf16(a1, b, acc[1][j]);
            }
        }
#pragma unroll
        for (int st = 0; st < 2; ++st) {
#pragma unroll
            for (int j = 0; j < 4; ++j) {
                int n0 = (colhalf * 4 + j) * 16;
#pragma unroll
                for (int r = 0; r < 8; ++r) {
                    int m = m0 + st * 16 + r + 8 * srow;
                    oy[(size_t)(row0 + m) * KIN + n0 + scol] = acc[st][j][r];
                }
            }
        }
    }
}

// ---------------- launcher ----------------
extern "C" void kernel_launch(void* const* d_in, const int* in_sizes, int n_in,
                              void* d_out, int out_size, void* d_ws, size_t ws_size,
                              hipStream_t stream) {
    // setup_inputs order:
    // 0 state1, 1 state2, 2 u, 3 A1, 4 B1, 5 C1, 6 D1, 7 A2, 8 B2, 9 C2, 10 D2,
    // 11 W1, 12 b1, 13 W2, 14 b2
    const float* s1 = (const float*)d_in[0];
    const float* s2 = (const float*)d_in[1];
    const float* u  = (const float*)d_in[2];
    const float* b1 = (const float*)d_in[12];
    const float* b2 = (const float*)d_in[14];
    __bf16* wbf = (__bf16*)d_ws;  // needs W_TOTAL*2 ~= 1.75 MB of scratch

    struct Cvt { int idx; int off; int cnt; };
    const Cvt cv[10] = {
        {3,  OFF_A1, KSD * KSD}, {4,  OFF_B1, KSD * KIN},
        {5,  OFF_C1, KSD * KSD}, {6,  OFF_D1, KSD * KIN},
        {7,  OFF_A2, KSD * KSD}, {8,  OFF_B2, KSD * KSD},
        {9,  OFF_C2, KIN * KSD}, {10, OFF_D2, KIN * KSD},
        {11, OFF_W1, KNH * KSD}, {13, OFF_W2, KSD * KNH},
    };
    for (int i = 0; i < 10; ++i) {
        wh_cvt_f32_bf16<<<(cv[i].cnt + 255) / 256, 256, 0, stream>>>(
            (const float*)d_in[cv[i].idx], wbf + cv[i].off, cv[i].cnt);
    }

    // 96KB dynamic LDS per workgroup (WGP has 320KB; ISA allows 320KB/WG)
    (void)hipFuncSetAttribute((const void*)wh_fused,
                              hipFuncAttributeMaxDynamicSharedMemorySize, SMEM_BYTES);

    float* ons1 = (float*)d_out;
    float* ons2 = ons1 + (size_t)KBATCH * KSD;
    float* oy   = ons2 + (size_t)KBATCH * KSD;

    wh_fused<<<KBATCH / KROWS, KTHREADS, SMEM_BYTES, stream>>>(
        s1, s2, u, b1, b2, wbf, ons1, ons2, oy);
}
